// MambaClassifier_77945066487977
// MI455X (gfx1250) — matrix operand order repaired and checked by reading, not compiled, not run on hardware
//
#include <hip/hip_runtime.h>
#include <hip/hip_bf16.h>

// ---------------- model constants ----------------
#define D_MODEL  1024
#define N_LAYERS 4
#define D_INNER  2048      // 2*D_MODEL
#define D_STATE  16
#define D_CONV   4
#define DT_RANK  64        // D_MODEL/16
#define BATCH    8
#define SEQ      512
#define ROWS     (BATCH * SEQ)     // 4096 token rows
#define DBC_DIM  (DT_RANK + 2 * D_STATE)   // 96

typedef __bf16 bf16;
typedef __attribute__((ext_vector_type(16))) __bf16 v16bf;
typedef __attribute__((ext_vector_type(8)))  __bf16 v8bf;
typedef __attribute__((ext_vector_type(8)))  float  v8f;

typedef __attribute__((ext_vector_type(4))) unsigned int u32x4;
typedef __attribute__((ext_vector_type(8))) int          i32x8;
typedef __attribute__((ext_vector_type(4))) int          i32x4;

#if __has_builtin(__builtin_amdgcn_tensor_load_to_lds) && \
    __has_builtin(__builtin_amdgcn_s_wait_tensorcnt)
#define HAVE_TDM 1
#else
#define HAVE_TDM 0
#endif

// float -> bf16, round-to-nearest-even, pure integer path
__device__ __forceinline__ bf16 f2b(float f) {
    union { float f; unsigned u; } a; a.f = f;
    unsigned r = a.u + 0x7FFFu + ((a.u >> 16) & 1u);
    union { unsigned short s; bf16 b; } o;
    o.s = (unsigned short)(r >> 16);
    return o.b;
}

__device__ __forceinline__ float silu_f(float x) {
    return x / (1.0f + __expf(-x));
}

// ---------------------------------------------------------------------------
// bf16 WMMA GEMM:  D[M,N] = A[M,K] * W[N,K]^T   (f32 accumulate)
// A rows and W rows are K-contiguous. M,N multiples of 16; K multiple of 32.
// Each wave owns FOUR 16x16 N-tiles sharing one A fragment (4 WMMAs per A
// load); 8 waves/block -> 16(M) x 512(N) strip per block.
// Ragged N (only the N=96 x_proj call) is handled *branchlessly*: inactive
// tiles read dummy B rows (valid memory) and compute throwaway WMMAs; only
// the stores are guarded. Keeps the K loop free of SALU control flow so
// loads and WMMAs software-pipeline, and EXEC is trivially all-ones at every
// WMMA (ISA 7.12 requirement).
// ---------------------------------------------------------------------------
__global__ __launch_bounds__(256)
void gemm_bf16_wmma(const bf16* __restrict__ A, const bf16* __restrict__ W,
                    float* __restrict__ D, int M, int N, int K) {
    const int lane = threadIdx.x & 31;
    const int wave = __builtin_amdgcn_readfirstlane((int)(threadIdx.x >> 5));
    const int half = lane >> 4;
    const int l16  = lane & 15;
    const int mtile = blockIdx.y * 16;
    const int nbase = blockIdx.x * 512 + wave * 64;     // 4 tiles of 16
    if (nbase >= N) return;                              // wave-uniform guard

    const size_t arow = (size_t)(mtile + l16) * K;
    bool   act[4];
    size_t brow[4];
    #pragma unroll
    for (int j = 0; j < 4; ++j) {
        act[j]  = (nbase + 16 * j) < N;
        // inactive tiles: clamp to W row `l16` (valid memory, result discarded)
        brow[j] = (size_t)((act[j] ? nbase + 16 * j : 0) + l16) * K;
    }

    v8f acc[4] = {{}, {}, {}, {}};
    for (int kb = 0; kb < K; kb += 32) {
        // ---- A fragment: two contiguous 8-element bf16 chunks per lane ----
        v8bf c0 = *(const v8bf*)(A + arow + kb + half * 8);
        v8bf c1 = *(const v8bf*)(A + arow + kb + 16 + half * 8);
        v16bf af;
        #pragma unroll
        for (int i = 0; i < 8; ++i) { af[i] = c0[i]; af[i + 8] = c1[i]; }
        __builtin_prefetch(A + arow + kb + 256, 0, 3);     // global_prefetch_b8
        __builtin_prefetch(W + brow[0] + kb + 256, 0, 3);
        #pragma unroll
        for (int j = 0; j < 4; ++j) {
            v8bf b0 = *(const v8bf*)(W + brow[j] + kb + half * 8);
            v8bf b1 = *(const v8bf*)(W + brow[j] + kb + 16 + half * 8);
            v16bf bfrag;
            #pragma unroll
            for (int i = 0; i < 8; ++i) { bfrag[i] = b0[i]; bfrag[i + 8] = b1[i]; }
            acc[j] = __builtin_amdgcn_wmma_f32_16x16x32_bf16(
                         false, af, false, bfrag, (short)0, acc[j],
                         false, false);
        }
    }
    #pragma unroll
    for (int j = 0; j < 4; ++j) {
        if (act[j]) {
            #pragma unroll
            for (int v = 0; v < 8; ++v) {
                int row = mtile + v + 8 * half;
                D[(size_t)row * N + nbase + 16 * j + l16] = acc[j][v];
            }
        }
    }
}

// ---------------------------------------------------------------------------
// Elementwise / support kernels
// ---------------------------------------------------------------------------
__global__ void embed_kernel(const int* __restrict__ ids,
                             const float* __restrict__ emb,
                             float* __restrict__ x) {
    size_t i = (size_t)blockIdx.x * blockDim.x + threadIdx.x;
    if (i >= (size_t)ROWS * D_MODEL) return;
    size_t row = i / D_MODEL, c = i % D_MODEL;
    x[i] = emb[(size_t)ids[row] * D_MODEL + c];
}

__global__ void f32_to_bf16_kernel(const float* __restrict__ src,
                                   bf16* __restrict__ dst, size_t n) {
    size_t i = (size_t)blockIdx.x * blockDim.x + threadIdx.x;
    if (i < n) dst[i] = f2b(src[i]);
}

// causal depthwise conv over time + bias + SiLU; reads xm-preact from XR
// (row-major [ROWS, 2*D_INNER], first D_INNER columns), writes fp32 + bf16.
__global__ void conv_silu_kernel(const float* __restrict__ xr,
                                 const float* __restrict__ cw,
                                 const float* __restrict__ cb,
                                 float* __restrict__ xm,
                                 bf16* __restrict__ xmb) {
    size_t i = (size_t)blockIdx.x * blockDim.x + threadIdx.x;
    if (i >= (size_t)ROWS * D_INNER) return;
    int d = (int)(i % D_INNER);
    size_t row = i / D_INNER;
    int l = (int)(row % SEQ);
    float acc = cb[d];
    #pragma unroll
    for (int k = 0; k < D_CONV; ++k) {
        int dt = k - (D_CONV - 1);           // -3..0
        if (l + dt >= 0)
            acc += cw[d * D_CONV + k] * xr[(row + dt) * (2 * D_INNER) + d];
    }
    float v = silu_f(acc);
    xm[i]  = v;
    xmb[i] = f2b(v);
}

// pull dt (first 64 cols of dbc) out as bf16 rows for the dt_proj GEMM
__global__ void dt_extract_kernel(const float* __restrict__ dbc,
                                  bf16* __restrict__ dtb) {
    size_t i = (size_t)blockIdx.x * blockDim.x + threadIdx.x;
    if (i >= (size_t)ROWS * DT_RANK) return;
    size_t row = i / DT_RANK, j = i % DT_RANK;
    dtb[i] = f2b(dbc[row * DBC_DIM + j]);
}

// delta = softplus(delta_raw + dt_bias[d])
__global__ void softplus_kernel(float* __restrict__ delta,
                                const float* __restrict__ bias) {
    size_t i = (size_t)blockIdx.x * blockDim.x + threadIdx.x;
    if (i >= (size_t)ROWS * D_INNER) return;
    int d = (int)(i % D_INNER);
    float x = delta[i] + bias[d];
    delta[i] = (x > 20.0f) ? x : log1pf(__expf(x));
}

#if HAVE_TDM
// Issue a TDM load of one 16-row x 32-float tile (row stride 96 floats) from
// global into LDS at byte offset `ldsoff`. D# built per CDNA5 ISA §8.3/8.4:
//   group0: count=1 | lds_addr | global_addr(57b) | type=2
//   group1: data_size=4B, tensor_dim0=32, tensor_dim1=16, tile=32x16,
//           tensor_dim0_stride=96 elements, no multicast/pad/iterate.
__device__ __forceinline__ void tdm_load_tile(unsigned ldsoff,
                                              const float* gptr) {
    unsigned long long ga = (unsigned long long)gptr;
    u32x4 g0;
    g0[0] = 1u;                                            // count=1 (valid)
    g0[1] = ldsoff;                                        // lds_addr [63:32]
    g0[2] = (unsigned)(ga & 0xFFFFFFFFull);                // gaddr [95:64]
    g0[3] = (unsigned)((ga >> 32) & 0x01FFFFFFull)         // gaddr [120:96]
            | 0x80000000u;                                 // type=2 [127:126]
    i32x8 g1;
    g1[0] = 0x00020000;        // workgroup_mask=0, data_size=2 (4 bytes)
    g1[1] = (32 << 16);        // tensor_dim0 lo16 at [31:16]
    g1[2] = (16 << 16);        // tensor_dim0 hi=0 | tensor_dim1 lo=16
    g1[3] = (32 << 16);        // tensor_dim1 hi=0 | tile_dim0=32
    g1[4] = 16;                // tile_dim1=16, tile_dim2=0
    g1[5] = 96;                // tensor_dim0_stride lo32 = 96 elements
    g1[6] = 0;                 // stride0 hi / tensor_dim1_stride lo
    g1[7] = 0;
    i32x4 z4 = {0, 0, 0, 0};
#if __clang_major__ >= 23
    i32x8 z8 = {0, 0, 0, 0, 0, 0, 0, 0};
    __builtin_amdgcn_tensor_load_to_lds(g0, g1, z4, z4, z8, 0);
#else
    __builtin_amdgcn_tensor_load_to_lds(g0, g1, z4, z4, 0);
#endif
}
#endif

// Selective scan. One lane per (b,d) channel; 16 states live in VGPRs.
// B/C vectors (shared by all 2048 channels of a batch) staged into LDS by the
// Tensor Data Mover, double-buffered: TDM fetches chunk t+16 while the wave
// grinds the 16 serial recurrence steps of chunk t (s_wait_tensorcnt sync).
//
// The TDM builtin carries no LDS pointer the compiler can alias with sBC:
//  - GlobalOpt folds loads of a store-free LDS global to undef (rounds 2/3
//    showed ds=0)  -> defeat with a dynamically-dead store guarded by a
//    runtime value (blockDim.x==0) the compiler cannot disprove.
//  - A volatile *generic* pointer forces flat loads, not ds loads (round 4
//    showed vmem up, ds still 0)  -> read sBC directly so clang keeps
//    addrspace(3) and emits ds_load.
//  - Per-iteration asm with the escaped LDS address + memory clobber models
//    the TDM writes and orders the reads after s_wait_tensorcnt.
// ys written into the (dead) xm half of XR, stride 4096.
__global__ __launch_bounds__(256)
void scan_kernel(const float* __restrict__ delta,
                 const float* __restrict__ xm,
                 const float* __restrict__ dbc,
                 const float* __restrict__ alog,
                 float* __restrict__ ys) {
    const int b = blockIdx.x / (D_INNER / 256);
    const int d = (blockIdx.x % (D_INNER / 256)) * 256 + threadIdx.x;

    float Areg[D_STATE], h[D_STATE];
    #pragma unroll
    for (int n = 0; n < D_STATE; ++n) {
        Areg[n] = -__expf(alog[(size_t)d * D_STATE + n]);
        h[n] = 0.0f;
    }

    // [buf][t][ B(0..15) | C(16..31) ] ; sole LDS object -> offset 0
    __shared__ float sBC[2][16][2 * D_STATE];

#if HAVE_TDM
    // Dynamically dead (blockDim.x==256) but not provably so at compile time:
    // gives sBC a visible store so loads cannot be folded to undef.
    if (blockDim.x == 0u) sBC[0][0][0] = 1.0f;

    float* lds_base = &sBC[0][0][0];
    const bool issuer = (threadIdx.x < 32);      // wave 0 drives the TDM
    if (issuer)
        tdm_load_tile(0u, dbc + (size_t)b * SEQ * DBC_DIM + DT_RANK);
    for (int t0 = 0; t0 < SEQ; t0 += 16) {
        const int buf = (t0 >> 4) & 1;
        if (issuer) __builtin_amdgcn_s_wait_tensorcnt(0);
        __syncthreads();                         // tile[buf] visible to all
        // escape LDS address into a may-write instruction: models TDM stores
        asm volatile("" : : "r"(lds_base) : "memory");
        if (issuer && (t0 + 16) < SEQ)           // prefetch next chunk
            tdm_load_tile((unsigned)((buf ^ 1) * 16 * 2 * D_STATE * 4),
                          dbc + ((size_t)b * SEQ + t0 + 16) * DBC_DIM + DT_RANK);
        for (int tt = 0; tt < 16; ++tt) {
            size_t row = (size_t)b * SEQ + t0 + tt;
            float dlt = delta[row * D_INNER + d];
            float xv  = xm[row * D_INNER + d];
            float dx  = dlt * xv;
            float y = 0.0f;
            #pragma unroll
            for (int n = 0; n < D_STATE; ++n) {
                h[n] = __expf(dlt * Areg[n]) * h[n] + dx * sBC[buf][tt][n];
                y += h[n] * sBC[buf][tt][D_STATE + n];
            }
            ys[row * (2 * D_INNER) + d] = y;
        }
        __syncthreads();
    }
#else
    for (int t0 = 0; t0 < SEQ; t0 += 16) {
        for (int j = threadIdx.x; j < 16 * 2 * D_STATE; j += 256) {
            int tt = j >> 5, e = j & 31;
            sBC[0][tt][e] =
                dbc[((size_t)b * SEQ + t0 + tt) * DBC_DIM + DT_RANK + e];
        }
        __syncthreads();
        for (int tt = 0; tt < 16; ++tt) {
            size_t row = (size_t)b * SEQ + t0 + tt;
            float dlt = delta[row * D_INNER + d];
            float xv  = xm[row * D_INNER + d];
            float dx  = dlt * xv;
            float y = 0.0f;
            #pragma unroll
            for (int n = 0; n < D_STATE; ++n) {
                h[n] = __expf(dlt * Areg[n]) * h[n] + dx * sBC[0][tt][n];
                y += h[n] * sBC[0][tt][D_STATE + n];
            }
            ys[row * (2 * D_INNER) + d] = y;
        }
        __syncthreads();
    }
#endif
}

// y = (ys + D[d]*xm) * silu(res); res lives in second half of XR rows.
__global__ void gate_kernel(const float* __restrict__ xrbuf,
                            const float* __restrict__ xm,
                            const float* __restrict__ Dp,
                            bf16* __restrict__ yb) {
    size_t i = (size_t)blockIdx.x * blockDim.x + threadIdx.x;
    if (i >= (size_t)ROWS * D_INNER) return;
    int d = (int)(i % D_INNER);
    size_t row = i / D_INNER;
    float y = xrbuf[row * (2 * D_INNER) + d] + Dp[d] * xm[i];
    float r = xrbuf[row * (2 * D_INNER) + D_INNER + d];
    yb[i] = f2b(y * silu_f(r));
}

__global__ void pool_kernel(const float* __restrict__ x,
                            float* __restrict__ pooled) {
    int i = blockIdx.x * blockDim.x + threadIdx.x;
    if (i >= BATCH * D_MODEL) return;
    int b = i / D_MODEL, c = i % D_MODEL;
    float s = 0.0f;
    for (int l = 0; l < SEQ; ++l)
        s += x[((size_t)b * SEQ + l) * D_MODEL + c];
    pooled[i] = s * (1.0f / SEQ);
}

__global__ void cls1_kernel(const float* __restrict__ pooled,
                            const float* __restrict__ w1,
                            const float* __restrict__ b1,
                            float* __restrict__ h1) {
    int i = blockIdx.x * blockDim.x + threadIdx.x;
    if (i >= BATCH * (D_MODEL / 2)) return;
    int b = i / (D_MODEL / 2), j = i % (D_MODEL / 2);
    float s = b1[j];
    for (int k = 0; k < D_MODEL; ++k)
        s += pooled[b * D_MODEL + k] * w1[(size_t)j * D_MODEL + k];
    h1[i] = s > 0.0f ? s : 0.0f;
}

__global__ void cls2_kernel(const float* __restrict__ h1,
                            const float* __restrict__ w2,
                            const float* __restrict__ b2,
                            float* __restrict__ out) {
    int i = blockIdx.x * blockDim.x + threadIdx.x;
    if (i >= BATCH * 10) return;
    int b = i / 10, c = i % 10;
    float s = b2[c];
    for (int k = 0; k < D_MODEL / 2; ++k)
        s += h1[b * (D_MODEL / 2) + k] * w2[(size_t)c * (D_MODEL / 2) + k];
    out[i] = s;
}

// ---------------------------------------------------------------------------
extern "C" void kernel_launch(void* const* d_in, const int* in_sizes, int n_in,
                              void* d_out, int out_size, void* d_ws, size_t ws_size,
                              hipStream_t stream) {
    (void)in_sizes; (void)n_in; (void)out_size; (void)ws_size;

    const int*   ids  = (const int*)  d_in[0];
    const float* emb  = (const float*)d_in[1];
    const float* inw  = (const float*)d_in[2];   // (4, 4096, 1024)
    const float* cw   = (const float*)d_in[3];   // (4, 2048, 4)
    const float* cb   = (const float*)d_in[4];   // (4, 2048)
    const float* xpw  = (const float*)d_in[5];   // (4, 96, 2048)
    const float* dtw  = (const float*)d_in[6];   // (4, 2048, 64)
    const float* dtb  = (const float*)d_in[7];   // (4, 2048)
    const float* alog = (const float*)d_in[8];   // (4, 2048, 16)
    const float* Dp   = (const float*)d_in[9];   // (4, 2048)
    const float* ow   = (const float*)d_in[10];  // (4, 1024, 2048)
    const float* w1   = (const float*)d_in[11];
    const float* b1   = (const float*)d_in[12];
    const float* w2   = (const float*)d_in[13];
    const float* b2   = (const float*)d_in[14];
    float* out = (float*)d_out;

    // ---- workspace carve (256B aligned slices) ----
    char* ws = (char*)d_ws;
    size_t off = 0;
    auto carve = [&](size_t bytes) -> void* {
        void* p = ws + off;
        off += (bytes + 255) & ~(size_t)255;
        return p;
    };
    float* X    = (float*)carve((size_t)ROWS * D_MODEL * 4);       // activations
    bf16*  XB   = (bf16*) carve((size_t)ROWS * D_MODEL * 2);
    float* XR   = (float*)carve((size_t)ROWS * 2 * D_INNER * 4);   // in_proj out; ys reuses col 0..2047
    float* XM   = (float*)carve((size_t)ROWS * D_INNER * 4);
    bf16*  XMB  = (bf16*) carve((size_t)ROWS * D_INNER * 2);
    float* DBC  = (float*)carve((size_t)ROWS * DBC_DIM * 4);
    bf16*  DTB  = (bf16*) carve((size_t)ROWS * DT_RANK * 2);
    float* DEL  = (float*)carve((size_t)ROWS * D_INNER * 4);
    bf16*  YB   = (bf16*) carve((size_t)ROWS * D_INNER * 2);
    bf16*  WIN  = (bf16*) carve((size_t)2 * D_INNER * D_MODEL * 2);
    bf16*  WXP  = (bf16*) carve((size_t)DBC_DIM * D_INNER * 2);
    bf16*  WDT  = (bf16*) carve((size_t)D_INNER * DT_RANK * 2);
    bf16*  WOUT = (bf16*) carve((size_t)D_MODEL * D_INNER * 2);
    float* POOL = (float*)carve(BATCH * D_MODEL * 4);
    float* H1   = (float*)carve(BATCH * (D_MODEL / 2) * 4);

    auto nblk  = [](size_t n) { return (unsigned)((n + 255) / 256); };
    auto ngrid = [](int N)    { return (unsigned)((N + 511) / 512); };

    // ---- embedding gather ----
    embed_kernel<<<nblk((size_t)ROWS * D_MODEL), 256, 0, stream>>>(ids, emb, X);

    for (int l = 0; l < N_LAYERS; ++l) {
        const float* inw_l  = inw  + (size_t)l * 2 * D_INNER * D_MODEL;
        const float* cw_l   = cw   + (size_t)l * D_INNER * D_CONV;
        const float* cb_l   = cb   + (size_t)l * D_INNER;
        const float* xpw_l  = xpw  + (size_t)l * DBC_DIM * D_INNER;
        const float* dtw_l  = dtw  + (size_t)l * D_INNER * DT_RANK;
        const float* dtb_l  = dtb  + (size_t)l * D_INNER;
        const float* alog_l = alog + (size_t)l * D_INNER * D_STATE;
        const float* Dp_l   = Dp   + (size_t)l * D_INNER;
        const float* ow_l   = ow   + (size_t)l * D_MODEL * D_INNER;

        // x -> bf16 ; in_proj weights -> bf16
        f32_to_bf16_kernel<<<nblk((size_t)ROWS * D_MODEL), 256, 0, stream>>>(X, XB, (size_t)ROWS * D_MODEL);
        f32_to_bf16_kernel<<<nblk((size_t)2 * D_INNER * D_MODEL), 256, 0, stream>>>(inw_l, WIN, (size_t)2 * D_INNER * D_MODEL);

        // xr = x @ in_w^T : M=4096 K=1024 N=4096
        gemm_bf16_wmma<<<dim3(ngrid(2 * D_INNER), ROWS / 16), 256, 0, stream>>>(
            XB, WIN, XR, ROWS, 2 * D_INNER, D_MODEL);

        // depthwise causal conv + SiLU
        conv_silu_kernel<<<nblk((size_t)ROWS * D_INNER), 256, 0, stream>>>(XR, cw_l, cb_l, XM, XMB);

        // dbc = xm @ x_proj^T : M=4096 K=2048 N=96
        f32_to_bf16_kernel<<<nblk((size_t)DBC_DIM * D_INNER), 256, 0, stream>>>(xpw_l, WXP, (size_t)DBC_DIM * D_INNER);
        gemm_bf16_wmma<<<dim3(ngrid(DBC_DIM), ROWS / 16), 256, 0, stream>>>(
            XMB, WXP, DBC, ROWS, DBC_DIM, D_INNER);

        // delta = softplus(dt @ dt_proj^T + dt_bias) : M=4096 K=64 N=2048
        dt_extract_kernel<<<nblk((size_t)ROWS * DT_RANK), 256, 0, stream>>>(DBC, DTB);
        f32_to_bf16_kernel<<<nblk((size_t)D_INNER * DT_RANK), 256, 0, stream>>>(dtw_l, WDT, (size_t)D_INNER * DT_RANK);
        gemm_bf16_wmma<<<dim3(ngrid(D_INNER), ROWS / 16), 256, 0, stream>>>(
            DTB, WDT, DEL, ROWS, D_INNER, DT_RANK);
        softplus_kernel<<<nblk((size_t)ROWS * D_INNER), 256, 0, stream>>>(DEL, dtb_l);

        // selective scan (ys -> XR first-half columns, stride 4096)
        scan_kernel<<<BATCH * (D_INNER / 256), 256, 0, stream>>>(DEL, XM, DBC, alog_l, XR);

        // gate: y = (ys + D*xm) * silu(res) -> bf16
        gate_kernel<<<nblk((size_t)ROWS * D_INNER), 256, 0, stream>>>(XR, XM, Dp_l, YB);

        // x = y @ out_proj^T : M=4096 K=2048 N=1024
        f32_to_bf16_kernel<<<nblk((size_t)D_MODEL * D_INNER), 256, 0, stream>>>(ow_l, WOUT, (size_t)D_MODEL * D_INNER);
        gemm_bf16_wmma<<<dim3(ngrid(D_MODEL), ROWS / 16), 256, 0, stream>>>(
            YB, WOUT, X, ROWS, D_MODEL, D_INNER);
    }

    // mean-pool over L, then 2-layer classifier head
    pool_kernel<<<nblk(BATCH * D_MODEL), 256, 0, stream>>>(X, POOL);
    cls1_kernel<<<nblk(BATCH * (D_MODEL / 2)), 256, 0, stream>>>(POOL, w1, b1, H1);
    cls2_kernel<<<1, 256, 0, stream>>>(H1, w2, b2, out);
}
